// DIEN_11836929868184
// MI455X (gfx1250) — matrix operand order repaired and checked by reading, not compiled, not run on hardware
//
#include <hip/hip_runtime.h>
#include <math.h>

// ---------------- CDNA5 WMMA plumbing (gfx1250, wave32) ----------------
typedef __bf16 bf16;
typedef __attribute__((ext_vector_type(16))) __bf16 v16bf;
typedef __attribute__((ext_vector_type(8)))  __bf16 v8bf;
typedef __attribute__((ext_vector_type(8)))  float  v8f;

#define B_      2048
#define L_      200
#define D_      128
#define G3D_    384
#define CONCAT_ 332
#define CPAD_   352   // CONCAT padded to multiple of 32 for WMMA K
#define MLP1_   256
#define MLP2_   128

__device__ __forceinline__ float sigmoidf_(float x) { return 1.0f / (1.0f + __expf(-x)); }

__device__ __forceinline__ v8f wmma_bf16_(v16bf a, v16bf b, v8f c) {
  // D = A(16x32 bf16) x B(32x16 bf16) + C(16x16 f32)
  return __builtin_amdgcn_wmma_f32_16x16x32_bf16(false, a, false, b, (short)0, c, false, false);
}

// A-fragment: row-major bf16 tile in LDS, base -> element [0][k0], row stride ld.
// A layout (matches B): lane m=l%16, half=l/16; elems 0-15 = K(16*half..16*half+15)
__device__ __forceinline__ v16bf load_a_lds(const bf16* base, int ld, int lane) {
  int m = lane & 15, h = lane >> 4;
  const bf16* r = base + m * ld + 16 * h;
  v8bf lo = *(const v8bf*)(r);
  v8bf hi = *(const v8bf*)(r + 8);
  v16bf a;
#pragma unroll
  for (int i = 0; i < 8; ++i) { a[i] = lo[i]; a[8 + i] = hi[i]; }
  return a;
}

// B-fragment for Y = X @ W^T with W (N,K) row-major bf16:
// B[k][n] = W[n][k]; lane n=l%16, half=l/16 holds K = k0+16*half .. +15 -> one contiguous 32B load.
__device__ __forceinline__ v16bf load_b_w(const bf16* w, int K, int n0, int k0, int lane) {
  int n = lane & 15, h = lane >> 4;
  return *(const v16bf*)(w + (size_t)(n0 + n) * K + k0 + 16 * h);
}

// ---------------- prep: bf16 weight conversion + h init ----------------
__global__ void prep_kernel(const float* gWi, const float* gWh,
                            const float* aWr, const float* aWu, const float* aWhW,
                            const float* mW1, const float* mW2, const float* tpW,
                            bf16* wWi, bf16* wWh, bf16* wAr, bf16* wAu, bf16* wAh,
                            bf16* wW1, bf16* wW2, bf16* wTP,
                            float* hf, bf16* hb) {
  int stride = gridDim.x * blockDim.x;
  int t0 = blockIdx.x * blockDim.x + threadIdx.x;
  for (int i = t0; i < G3D_ * D_; i += stride) { wWi[i] = (bf16)gWi[i]; wWh[i] = (bf16)gWh[i]; }
  for (int i = t0; i < D_ * 256; i += stride) {
    wAr[i] = (bf16)aWr[i]; wAu[i] = (bf16)aWu[i]; wAh[i] = (bf16)aWhW[i];
  }
  for (int i = t0; i < MLP1_ * CPAD_; i += stride) {
    int r = i / CPAD_, c = i % CPAD_;
    wW1[i] = (c < CONCAT_) ? (bf16)mW1[(size_t)r * CONCAT_ + c] : (bf16)0.f;
  }
  for (int i = t0; i < MLP2_ * MLP1_; i += stride) wW2[i] = (bf16)mW2[i];
  for (int i = t0; i < D_ * 64; i += stride) wTP[i] = (bf16)tpW[i];
  for (int i = t0; i < B_ * D_; i += stride) { hf[i] = 0.f; hb[i] = (bf16)0.f; }
}

// ---------------- zero hidden state (graph-capture-safe replacement for memset) ----------------
__global__ void hzero_kernel(float* hf, bf16* hb) {
  int stride = gridDim.x * blockDim.x;
  for (int i = blockIdx.x * blockDim.x + threadIdx.x; i < B_ * D_; i += stride) {
    hf[i] = 0.f;
    hb[i] = (bf16)0.f;
  }
}

// ---------------- concat static features into padded bf16 MLP input ----------------
__global__ void concat_kernel(const int* uid, const int* iid, const int* cat, const int* dur,
                              const float* ud, const float* idn,
                              const float* utab, const float* itab,
                              const float* ctab, const float* dtab, bf16* xcat) {
  int stride = gridDim.x * blockDim.x;
  for (int i = blockIdx.x * blockDim.x + threadIdx.x; i < B_ * CPAD_; i += stride) {
    int b = i / CPAD_, c = i % CPAD_;
    float v;
    if      (c < 64)  v = utab[(size_t)uid[b] * 64 + c];
    else if (c < 128) v = itab[(size_t)iid[b] * 64 + (c - 64)];
    else if (c < 160) v = ctab[(size_t)cat[b] * 32 + (c - 128)];
    else if (c < 176) v = dtab[(size_t)dur[b] * 16 + (c - 160)];
    else if (c < 201) v = ud[(size_t)b * 25 + (c - 176)];
    else if (c < 204) v = idn[(size_t)b * 3 + (c - 201)];
    else              v = 0.f;   // evolved (filled post-AUGRU) + zero pad
    xcat[i] = (bf16)v;
  }
}

// ---------------- target = item_emb @ target_proj_W^T  (WMMA) ----------------
__global__ __launch_bounds__(256) void tproj_kernel(const int* __restrict__ item_id,
                                                    const float* __restrict__ item_table,
                                                    const bf16* __restrict__ wTP,
                                                    float* __restrict__ target) {
  __shared__ bf16 la[16 * 64];
  int tid = threadIdx.x, r0 = blockIdx.x * 16;
  for (int i = tid; i < 16 * 64; i += 256) {
    int m = i >> 6, c = i & 63;
    la[i] = (bf16)item_table[(size_t)item_id[r0 + m] * 64 + c];
  }
  __syncthreads();
  int lane = tid & 31, w = tid >> 5;
  int n0 = w * 16;                    // 8 waves cover N=128
  v8f acc = {};
#pragma unroll
  for (int kt = 0; kt < 2; ++kt) {
    v16bf a = load_a_lds(la + kt * 32, 64, lane);
    acc = wmma_bf16_(a, load_b_w(wTP, 64, n0, kt * 32, lane), acc);
  }
  int n = lane & 15, hl = lane >> 4;
#pragma unroll
  for (int v = 0; v < 8; ++v)
    target[(size_t)(r0 + v + 8 * hl) * D_ + n0 + n] = acc[v];
}

// ---------------- GRU step t: gather x_t; gi = x@Wi^T, gh = h@Wh^T (WMMA); gates ----------------
__global__ __launch_bounds__(256) void gru_step_kernel(const int* __restrict__ hseq,
                                                       const float* __restrict__ hist_table,
                                                       const bf16* __restrict__ wWi,
                                                       const bf16* __restrict__ wWh,
                                                       const float* __restrict__ bi,
                                                       const float* __restrict__ bh,
                                                       float* __restrict__ hf,
                                                       bf16* __restrict__ hb,
                                                       bf16* __restrict__ hs, int t) {
  __shared__ bf16 lx[16 * D_];
  __shared__ bf16 lh[16 * D_];
  __shared__ float gi[16 * G3D_];
  __shared__ float gh[16 * G3D_];
  int tid = threadIdx.x, r0 = blockIdx.x * 16;
  for (int i = tid; i < 16 * D_; i += 256) {
    int m = i / D_, c = i % D_;
    int row = r0 + m;
    int idx = hseq[row * L_ + t];                       // row 0 of hist_table is the zero pad row
    lx[i] = (bf16)hist_table[(size_t)idx * D_ + c];
    lh[i] = hb[(size_t)row * D_ + c];
  }
  __syncthreads();
  int lane = tid & 31, w = tid >> 5;
#pragma unroll
  for (int nt = 0; nt < 3; ++nt) {                     // 8 waves x 3 tiles = N 0..383
    int n0 = (w + nt * 8) * 16;
    v8f ai = {}; v8f ah = {};
#pragma unroll
    for (int kt = 0; kt < 4; ++kt) {                   // K = 128
      v16bf ax = load_a_lds(lx + kt * 32, D_, lane);
      ai = wmma_bf16_(ax, load_b_w(wWi, D_, n0, kt * 32, lane), ai);
      v16bf ahh = load_a_lds(lh + kt * 32, D_, lane);
      ah = wmma_bf16_(ahh, load_b_w(wWh, D_, n0, kt * 32, lane), ah);
    }
    int n = lane & 15, hl = lane >> 4;
#pragma unroll
    for (int v = 0; v < 8; ++v) {
      gi[(v + 8 * hl) * G3D_ + n0 + n] = ai[v];
      gh[(v + 8 * hl) * G3D_ + n0 + n] = ah[v];
    }
  }
  __syncthreads();
  {
    int m = tid >> 4, c0 = (tid & 15) * 8;
    int row = r0 + m;
#pragma unroll
    for (int j = 0; j < 8; ++j) {
      int c = c0 + j;
      float r  = sigmoidf_(gi[m * G3D_ + c]       + bi[c]       + gh[m * G3D_ + c]       + bh[c]);
      float z  = sigmoidf_(gi[m * G3D_ + 128 + c] + bi[128 + c] + gh[m * G3D_ + 128 + c] + bh[128 + c]);
      float nn = tanhf(gi[m * G3D_ + 256 + c] + bi[256 + c] +
                       r * (gh[m * G3D_ + 256 + c] + bh[256 + c]));
      float hold = hf[(size_t)row * D_ + c];
      float hnew = (1.0f - z) * nn + z * hold;
      hf[(size_t)row * D_ + c] = hnew;
      bf16 hv = (bf16)hnew;
      hb[(size_t)row * D_ + c] = hv;
      hs[((size_t)row * L_ + t) * D_ + c] = hv;
    }
  }
}

// ---------------- masked softmax attention over L ----------------
__global__ __launch_bounds__(256) void attn_kernel(const bf16* __restrict__ hs,
                                                   const float* __restrict__ target,
                                                   const int* __restrict__ hseq,
                                                   float* __restrict__ attn) {
  __shared__ float tg[D_];
  __shared__ float red[256];
  int b = blockIdx.x, tid = threadIdx.x;
  if (tid < D_) tg[tid] = target[(size_t)b * D_ + tid];
  __syncthreads();
  float sc = -1e30f;
  if (tid < L_) {
    const bf16* hrow = hs + ((size_t)b * L_ + tid) * D_;
    float s = 0.f;
#pragma unroll 8
    for (int k = 0; k < D_; ++k) s += (float)hrow[k] * tg[k];
    s *= 0.08838834764831845f;                          // 1/sqrt(128)
    sc = (hseq[(size_t)b * L_ + tid] > 0) ? s : -1e9f;
  }
  red[tid] = sc;
  __syncthreads();
  for (int s = 128; s > 0; s >>= 1) {
    if (tid < s) red[tid] = fmaxf(red[tid], red[tid + s]);
    __syncthreads();
  }
  float mx = red[0];
  __syncthreads();
  float ex = (tid < L_) ? __expf(sc - mx) : 0.f;
  red[tid] = ex;
  __syncthreads();
  for (int s = 128; s > 0; s >>= 1) {
    if (tid < s) red[tid] += red[tid + s];
    __syncthreads();
  }
  if (tid < L_) attn[(size_t)b * L_ + tid] = ex / red[0];
}

// ---------------- AUGRU step t: phase1 r,u (WMMA), phase2 h_tilde (WMMA), blend ----------------
__global__ __launch_bounds__(256) void augru_step_kernel(const int* __restrict__ hseq,
                                                         const bf16* __restrict__ hs,
                                                         const float* __restrict__ attn,
                                                         const bf16* __restrict__ wAr,
                                                         const bf16* __restrict__ wAu,
                                                         const bf16* __restrict__ wAh,
                                                         const float* __restrict__ br,
                                                         const float* __restrict__ bu,
                                                         const float* __restrict__ bhv,
                                                         float* __restrict__ hf,
                                                         bf16* __restrict__ hb, int t) {
  __shared__ bf16  a1[16 * 256];   // [x_t | h]
  __shared__ bf16  a2[16 * 256];   // [x_t | r*h]
  __shared__ float ru[16 * 256];   // r in cols 0..127, u in 128..255
  __shared__ float ht[16 * 128];
  int tid = threadIdx.x, r0 = blockIdx.x * 16;
  for (int i = tid; i < 16 * 256; i += 256) {
    int m = i >> 8, c = i & 255;
    int row = r0 + m;
    a1[i] = (c < 128) ? hs[((size_t)row * L_ + t) * D_ + c]
                      : hb[(size_t)row * D_ + (c - 128)];
  }
  __syncthreads();
  int lane = tid & 31, w = tid >> 5;
  int n0 = w * 16;                                      // 8 waves cover N=128
  {
    v8f ar = {}; v8f au = {};
#pragma unroll
    for (int kt = 0; kt < 8; ++kt) {                    // K = 256
      v16bf a = load_a_lds(a1 + kt * 32, 256, lane);
      ar = wmma_bf16_(a, load_b_w(wAr, 256, n0, kt * 32, lane), ar);
      au = wmma_bf16_(a, load_b_w(wAu, 256, n0, kt * 32, lane), au);
    }
    int n = lane & 15, hl = lane >> 4;
#pragma unroll
    for (int v = 0; v < 8; ++v) {
      int m = v + 8 * hl;
      ru[m * 256 + n0 + n]       = sigmoidf_(ar[v] + br[n0 + n]);
      ru[m * 256 + 128 + n0 + n] = sigmoidf_(au[v] + bu[n0 + n]);
    }
  }
  __syncthreads();
  for (int i = tid; i < 16 * 256; i += 256) {
    int m = i >> 8, c = i & 255;
    int row = r0 + m;
    if (c < 128) a2[i] = a1[i];
    else a2[i] = (bf16)(ru[m * 256 + (c - 128)] * hf[(size_t)row * D_ + (c - 128)]);
  }
  __syncthreads();
  {
    v8f at = {};
#pragma unroll
    for (int kt = 0; kt < 8; ++kt) {
      v16bf a = load_a_lds(a2 + kt * 32, 256, lane);
      at = wmma_bf16_(a, load_b_w(wAh, 256, n0, kt * 32, lane), at);
    }
    int n = lane & 15, hl = lane >> 4;
#pragma unroll
    for (int v = 0; v < 8; ++v)
      ht[(v + 8 * hl) * 128 + n0 + n] = tanhf(at[v] + bhv[n0 + n]);
  }
  __syncthreads();
  {
    int m = tid >> 4, c0 = (tid & 15) * 8;
    int row = r0 + m;
    float e = attn[(size_t)row * L_ + t];
    bool valid = hseq[row * L_ + t] > 0;
#pragma unroll
    for (int j = 0; j < 8; ++j) {
      int c = c0 + j;
      float up   = e * ru[m * 256 + 128 + c];
      float hold = hf[(size_t)row * D_ + c];
      float hnew = (1.0f - up) * hold + up * ht[m * 128 + c];
      float outv = valid ? hnew : hold;
      hf[(size_t)row * D_ + c] = outv;
      hb[(size_t)row * D_ + c] = (bf16)outv;
    }
  }
}

// ---------------- evolved -> xcat slice ----------------
__global__ void evolved_kernel(const float* hf, bf16* xcat) {
  int stride = gridDim.x * blockDim.x;
  for (int i = blockIdx.x * blockDim.x + threadIdx.x; i < B_ * D_; i += stride) {
    int b = i / D_, c = i % D_;
    xcat[(size_t)b * CPAD_ + 204 + c] = (bf16)hf[i];
  }
}

// ---------------- MLP layer 1: 2048x256x352 (WMMA) ----------------
__global__ __launch_bounds__(256) void mlp1_kernel(const bf16* __restrict__ xcat,
                                                   const bf16* __restrict__ w1,
                                                   const float* __restrict__ b1,
                                                   bf16* __restrict__ y1) {
  __shared__ bf16 la[16 * CPAD_];
  int tid = threadIdx.x, r0 = blockIdx.x * 16;
  for (int i = tid; i < 16 * CPAD_; i += 256) la[i] = xcat[(size_t)r0 * CPAD_ + i];
  __syncthreads();
  int lane = tid & 31, w = tid >> 5;
#pragma unroll
  for (int nt = 0; nt < 2; ++nt) {                      // 8 waves x 2 tiles = N 0..255
    int n0 = (w + nt * 8) * 16;
    v8f acc = {};
#pragma unroll
    for (int kt = 0; kt < CPAD_ / 32; ++kt) {
      v16bf a = load_a_lds(la + kt * 32, CPAD_, lane);
      acc = wmma_bf16_(a, load_b_w(w1, CPAD_, n0, kt * 32, lane), acc);
    }
    int n = lane & 15, hl = lane >> 4;
#pragma unroll
    for (int v = 0; v < 8; ++v) {
      float o = acc[v] + b1[n0 + n];
      y1[(size_t)(r0 + v + 8 * hl) * MLP1_ + n0 + n] = (bf16)fmaxf(o, 0.f);
    }
  }
}

// ---------------- MLP layer 2: 2048x128x256 (WMMA) ----------------
__global__ __launch_bounds__(256) void mlp2_kernel(const bf16* __restrict__ y1,
                                                   const bf16* __restrict__ w2,
                                                   const float* __restrict__ b2,
                                                   bf16* __restrict__ y2) {
  __shared__ bf16 la[16 * MLP1_];
  int tid = threadIdx.x, r0 = blockIdx.x * 16;
  for (int i = tid; i < 16 * MLP1_; i += 256) la[i] = y1[(size_t)r0 * MLP1_ + i];
  __syncthreads();
  int lane = tid & 31, w = tid >> 5;
  int n0 = w * 16;
  v8f acc = {};
#pragma unroll
  for (int kt = 0; kt < MLP1_ / 32; ++kt) {
    v16bf a = load_a_lds(la + kt * 32, MLP1_, lane);
    acc = wmma_bf16_(a, load_b_w(w2, MLP1_, n0, kt * 32, lane), acc);
  }
  int n = lane & 15, hl = lane >> 4;
#pragma unroll
  for (int v = 0; v < 8; ++v) {
    float o = acc[v] + b2[n0 + n];
    y2[(size_t)(r0 + v + 8 * hl) * MLP2_ + n0 + n] = (bf16)fmaxf(o, 0.f);
  }
}

// ---------------- head: logit + sigmoid ----------------
__global__ void head_kernel(const bf16* __restrict__ y2, const float* __restrict__ w3,
                            const float* __restrict__ b3, float* __restrict__ out) {
  int row = blockIdx.x * blockDim.x + threadIdx.x;
  if (row >= B_) return;
  float acc = b3[0];
#pragma unroll 8
  for (int k = 0; k < MLP2_; ++k) acc += (float)y2[(size_t)row * MLP2_ + k] * w3[k];
  out[row] = 1.0f / (1.0f + __expf(-acc));
}

// ---------------- host launcher ----------------
extern "C" void kernel_launch(void* const* d_in, const int* in_sizes, int n_in,
                              void* d_out, int out_size, void* d_ws, size_t ws_size,
                              hipStream_t stream) {
  (void)in_sizes; (void)n_in; (void)out_size; (void)ws_size;
  const int*   uid    = (const int*)d_in[0];
  const int*   iid    = (const int*)d_in[1];
  const int*   cat    = (const int*)d_in[2];
  const int*   dur    = (const int*)d_in[3];
  const int*   hseq   = (const int*)d_in[4];
  const float* udense = (const float*)d_in[5];
  const float* idense = (const float*)d_in[6];
  const float* utab   = (const float*)d_in[7];
  const float* itab   = (const float*)d_in[8];
  const float* ctab   = (const float*)d_in[9];
  const float* dtab   = (const float*)d_in[10];
  const float* htab   = (const float*)d_in[11];
  const float* tpW    = (const float*)d_in[12];
  const float* gWi    = (const float*)d_in[13];
  const float* gWh    = (const float*)d_in[14];
  const float* gbi    = (const float*)d_in[15];
  const float* gbh    = (const float*)d_in[16];
  const float* aWr    = (const float*)d_in[17];
  const float* abr    = (const float*)d_in[18];
  const float* aWu    = (const float*)d_in[19];
  const float* abu    = (const float*)d_in[20];
  const float* aWh    = (const float*)d_in[21];
  const float* abh    = (const float*)d_in[22];
  const float* mW1    = (const float*)d_in[23];
  const float* mb1    = (const float*)d_in[24];
  const float* mW2    = (const float*)d_in[25];
  const float* mb2    = (const float*)d_in[26];
  const float* mW3    = (const float*)d_in[27];
  const float* mb3    = (const float*)d_in[28];
  float* out = (float*)d_out;

  char* ws = (char*)d_ws;
  size_t off = 0;
  auto carve = [&](size_t bytes) -> char* {
    char* p = ws + off;
    off += (bytes + 255) & ~(size_t)255;
    return p;
  };
  bf16*  wWi    = (bf16*)carve((size_t)G3D_ * D_ * 2);
  bf16*  wWh    = (bf16*)carve((size_t)G3D_ * D_ * 2);
  bf16*  wAr    = (bf16*)carve((size_t)D_ * 256 * 2);
  bf16*  wAu    = (bf16*)carve((size_t)D_ * 256 * 2);
  bf16*  wAhp   = (bf16*)carve((size_t)D_ * 256 * 2);
  bf16*  wW1    = (bf16*)carve((size_t)MLP1_ * CPAD_ * 2);
  bf16*  wW2    = (bf16*)carve((size_t)MLP2_ * MLP1_ * 2);
  bf16*  wTP    = (bf16*)carve((size_t)D_ * 64 * 2);
  float* hf     = (float*)carve((size_t)B_ * D_ * 4);
  bf16*  hb     = (bf16*)carve((size_t)B_ * D_ * 2);
  bf16*  hstate = (bf16*)carve((size_t)B_ * L_ * D_ * 2);   // 100 MB
  float* attn   = (float*)carve((size_t)B_ * L_ * 4);
  float* target = (float*)carve((size_t)B_ * D_ * 4);
  bf16*  xcat   = (bf16*)carve((size_t)B_ * CPAD_ * 2);
  bf16*  y1     = (bf16*)carve((size_t)B_ * MLP1_ * 2);
  bf16*  y2     = (bf16*)carve((size_t)B_ * MLP2_ * 2);

  prep_kernel<<<512, 256, 0, stream>>>(gWi, gWh, aWr, aWu, aWh, mW1, mW2, tpW,
                                       wWi, wWh, wAr, wAu, wAhp, wW1, wW2, wTP, hf, hb);
  concat_kernel<<<512, 256, 0, stream>>>(uid, iid, cat, dur, udense, idense,
                                         utab, itab, ctab, dtab, xcat);
  tproj_kernel<<<B_ / 16, 256, 0, stream>>>(iid, itab, wTP, target);

  for (int t = 0; t < L_; ++t)
    gru_step_kernel<<<B_ / 16, 256, 0, stream>>>(hseq, htab, wWi, wWh, gbi, gbh,
                                                 hf, hb, hstate, t);

  attn_kernel<<<B_, 256, 0, stream>>>(hstate, target, hseq, attn);

  // AUGRU h0 = 0 (pure kernel node, graph-capture safe)
  hzero_kernel<<<512, 256, 0, stream>>>(hf, hb);

  for (int t = 0; t < L_; ++t)
    augru_step_kernel<<<B_ / 16, 256, 0, stream>>>(hseq, hstate, attn,
                                                   wAr, wAu, wAhp, abr, abu, abh,
                                                   hf, hb, t);

  evolved_kernel<<<512, 256, 0, stream>>>(hf, xcat);
  mlp1_kernel<<<B_ / 16, 256, 0, stream>>>(xcat, wW1, mb1, y1);
  mlp2_kernel<<<B_ / 16, 256, 0, stream>>>(y1, wW2, mb2, y2);
  head_kernel<<<(B_ + 255) / 256, 256, 0, stream>>>(y2, mW3, mb3, out);
}
